// Attention_36163624632388
// MI455X (gfx1250) — matrix operand order repaired and checked by reading, not compiled, not run on hardware
//
#include <hip/hip_runtime.h>
#include <hip/hip_bf16.h>
#include <math.h>

// ---------------------------------------------------------------------------
// Types for CDNA5 WMMA (wave32): 16x16x32 f16 -> f32
// ---------------------------------------------------------------------------
typedef __attribute__((ext_vector_type(16))) _Float16 v16h;
typedef __attribute__((ext_vector_type(8)))  _Float16 v8h;
typedef __attribute__((ext_vector_type(4)))  _Float16 v4h;
typedef __attribute__((ext_vector_type(8)))  float    v8f;
typedef __attribute__((ext_vector_type(4)))  unsigned int u32x4;
typedef __attribute__((ext_vector_type(8)))  int      i32x8;
typedef __attribute__((ext_vector_type(4)))  int      i32x4;

#define WMMA_F16(a, b, c) \
  __builtin_amdgcn_wmma_f32_16x16x32_f16(false, (a), false, (b), (short)0, (c), false, false)

__device__ inline v16h cat8(v8h a, v8h b) {
  return __builtin_shufflevector(a, b, 0, 1, 2, 3, 4, 5, 6, 7, 8, 9, 10, 11, 12, 13, 14, 15);
}

// A-fragment: 16(M) x 32(K) f16, row-major source (stride in elements).
// Per ISA 7.12.2: lanes 0-15 hold K=0..7 & 16..23 of row M=lane;
// lanes 16-31 hold K=8..15 & 24..31.
__device__ inline v16h fragA(const _Float16* t, int stride) {
  const int lane = threadIdx.x & 31, ln = lane & 15, hi = lane >> 4;
  const _Float16* p = t + ln * stride;
  v8h lo = *(const v8h*)(p + hi * 8);
  v8h hh = *(const v8h*)(p + 16 + hi * 8);
  return cat8(lo, hh);
}

// B-fragment: 32(K) x 16(N) f16 where source row n holds K-contiguous data
// (i.e. B^T row-major). Lanes 0-15: N=lane, K=0..7 & 16..23; lanes 16-31: K=8..15 & 24..31.
__device__ inline v16h fragB(const _Float16* t, int stride) {
  const int lane = threadIdx.x & 31, ln = lane & 15, hi = lane >> 4;
  const _Float16* p = t + ln * stride;
  return cat8(*(const v8h*)(p + hi * 8), *(const v8h*)(p + 16 + hi * 8));
}

// ---------------------------------------------------------------------------
// Tensor Data Mover: 2D tile load (f16 elements) global -> LDS.
// Hardware pads the LDS destination 16B after every 64B row chunk, which
// reproduces the 40-element padded row stride the fragment loaders expect.
//   tile_d0  : contiguous elements per row (32 here -> 64B = 16 DWORDs)
//   tile_d1  : number of rows (128)
//   stride0  : tensor dim-0 stride in elements (row pitch, K)
// ---------------------------------------------------------------------------
__device__ inline void tdm_load_tile_f16(unsigned lds_byte_addr, const void* gaddr,
                                         int tensor_d0, int tensor_d1, int stride0,
                                         int tile_d0, int tile_d1) {
  const unsigned long long ga = (unsigned long long)gaddr;
  u32x4 g0;
  g0[0] = 1u;                                            // count=1, user descriptor
  g0[1] = lds_byte_addr;                                 // D#.lds_addr
  g0[2] = (unsigned)(ga & 0xFFFFFFFFu);                  // global_addr[31:0]
  g0[3] = (unsigned)((ga >> 32) & 0x01FFFFFFu) | (2u << 30);  // addr[56:32] | type=2
  i32x8 g1;
  g1[0] = (1 << 16)        // data_size = 1 (2 bytes)
        | (1 << 20)        // pad_enable
        | (3 << 22)        // pad_interval: every 16 DWORDs (64B)
        | (3 << 25);       // pad_amount: 4 DWORDs (16B)
  g1[1] = (tensor_d0 & 0xFFFF) << 16;                              // tensor_dim0[15:0]
  g1[2] = ((tensor_d0 >> 16) & 0xFFFF) | ((tensor_d1 & 0xFFFF) << 16);
  g1[3] = ((tensor_d1 >> 16) & 0xFFFF) | ((tile_d0 & 0xFFFF) << 16);
  g1[4] = (tile_d1 & 0xFFFF);                                      // tile_dim1 (tile_dim2=0)
  g1[5] = stride0;                                                 // tensor_dim0_stride[31:0]
  g1[6] = 0;                                                       // stride hi / dim1 stride
  g1[7] = 0;
  const i32x4 z4 = {0, 0, 0, 0};                                   // groups 2/3 unused (2D)
  const i32x8 z8 = {0, 0, 0, 0, 0, 0, 0, 0};
  __builtin_amdgcn_tensor_load_to_lds(g0, g1, z4, z4, z8, 0);
}

// ---------------------------------------------------------------------------
// fp32 -> fp16 conversion (x and the four weight matrices)
// ---------------------------------------------------------------------------
__global__ __launch_bounds__(256) void cvt_f32_f16(const float* __restrict__ s,
                                                   _Float16* __restrict__ d, int n) {
  int i = (blockIdx.x * 256 + threadIdx.x) * 4;
  if (i >= n) return;
  float4 f = *(const float4*)(s + i);
  v4h h = {(_Float16)f.x, (_Float16)f.y, (_Float16)f.z, (_Float16)f.w};
  *(v4h*)(d + i) = h;
}

// ---------------------------------------------------------------------------
// NT GEMM: C[m,n] = sum_k A[m*K+k] * B[n*K+k]   (einsum 'bsd,ed->bse')
// 256 threads = 8 waves; block tile 128x128, BK=32; wave tile 32x64.
// Tiles are DMA'd by the Tensor Data Mover (wave 0 issues, TENSORcnt
// tracked), double-buffered so tile k+1 is in flight during tile k's WMMAs.
// ---------------------------------------------------------------------------
template <typename TOUT>
__global__ __launch_bounds__(256) void gemm_nt(const _Float16* __restrict__ A,
                                               const _Float16* __restrict__ Bw,
                                               TOUT* __restrict__ C,
                                               int M, int N, int K) {
  __shared__ _Float16 sA[2][128 * 40];  // rows padded 32 -> 40 elems by TDM
  __shared__ _Float16 sB[2][128 * 40];
  const int n0 = blockIdx.x * 128, m0 = blockIdx.y * 128;
  const int tid = threadIdx.x, wid = tid >> 5, lane = tid & 31;
  const int ln = lane & 15, hi = lane >> 4;
  const int m_off = (wid & 3) * 32, n_off = (wid >> 2) * 64;

  const v8f vzero = {0.f, 0.f, 0.f, 0.f, 0.f, 0.f, 0.f, 0.f};
  v8f acc[2][4];
#pragma unroll
  for (int i = 0; i < 2; ++i)
#pragma unroll
    for (int j = 0; j < 4; ++j) acc[i][j] = vzero;

  const _Float16* gA = A + (long)m0 * K;   // block's 128 A rows
  const _Float16* gB = Bw + (long)n0 * K;  // block's 128 B rows
  const unsigned ldsA0 = (unsigned)(unsigned long long)&sA[0][0];
  const unsigned ldsA1 = (unsigned)(unsigned long long)&sA[1][0];
  const unsigned ldsB0 = (unsigned)(unsigned long long)&sB[0][0];
  const unsigned ldsB1 = (unsigned)(unsigned long long)&sB[1][0];
  const int NT = K >> 5;  // 32-wide K tiles

  if (wid == 0) {  // wave 0 issues DMA; EXEC!=0 only for wave 0
    tdm_load_tile_f16(ldsA0, gA, K, M, K, 32, 128);
    tdm_load_tile_f16(ldsB0, gB, K, N, K, 32, 128);
  }

  for (int kt = 0; kt < NT; ++kt) {
    const int cur = kt & 1;
    if (wid == 0) {
      if (kt + 1 < NT) {  // prefetch next tile into the other buffer
        const long koff = (long)(kt + 1) * 32;
        tdm_load_tile_f16(cur ? ldsA0 : ldsA1, gA + koff, K, M, K, 32, 128);
        tdm_load_tile_f16(cur ? ldsB0 : ldsB1, gB + koff, K, N, K, 32, 128);
        __builtin_amdgcn_s_wait_tensorcnt(2);  // tile kt landed; kt+1 in flight
      } else {
        __builtin_amdgcn_s_wait_tensorcnt(0);
      }
    }
    __syncthreads();  // publish tile kt to all waves

    v16h a0 = fragA(&sA[cur][(m_off)*40], 40);
    v16h a1 = fragA(&sA[cur][(m_off + 16) * 40], 40);
#pragma unroll
    for (int j = 0; j < 4; ++j) {
      v16h b = fragB(&sB[cur][(n_off + j * 16) * 40], 40);
      acc[0][j] = WMMA_F16(a0, b, acc[0][j]);
      acc[1][j] = WMMA_F16(a1, b, acc[1][j]);
    }
    __syncthreads();  // all reads of buffer `cur` done before it is re-filled
  }

  // C/D layout: lane(0-15)=N, vgpr v = M (lanes 16-31: M = v+8).
#pragma unroll
  for (int i = 0; i < 2; ++i)
#pragma unroll
    for (int j = 0; j < 4; ++j) {
      const int mb = m0 + m_off + i * 16 + hi * 8;
      const int nb = n0 + n_off + j * 16 + ln;
#pragma unroll
      for (int v = 0; v < 8; ++v)
        C[(long)(mb + v) * N + nb] = (TOUT)acc[i][j][v];
    }
}

// ---------------------------------------------------------------------------
// RoPE (in place, Q and K), half = 64, layout [B, S, H, 128]
// ---------------------------------------------------------------------------
__global__ __launch_bounds__(256) void rope_qk(_Float16* __restrict__ Q,
                                               _Float16* __restrict__ Kt,
                                               const float* __restrict__ cosc,
                                               const float* __restrict__ sinc,
                                               const int* __restrict__ offp) {
  const int idx = blockIdx.x * 256 + threadIdx.x;   // B*S*H*64 = 4,194,304
  const int d = idx & 63;
  const int h = (idx >> 6) & 15;
  const int s = (idx >> 10) & 2047;
  const int b = idx >> 21;
  const int pos = s + offp[0];
  const float c = cosc[pos * 64 + d];
  const float sn = sinc[pos * 64 + d];
  const long base = ((long)(b * 2048 + s)) * 2048 + h * 128 + d;
  float q1 = (float)Q[base], q2 = (float)Q[base + 64];
  Q[base]      = (_Float16)(q1 * c - q2 * sn);
  Q[base + 64] = (_Float16)(q1 * sn + q2 * c);
  float k1 = (float)Kt[base], k2 = (float)Kt[base + 64];
  Kt[base]      = (_Float16)(k1 * c - k2 * sn);
  Kt[base + 64] = (_Float16)(k1 * sn + k2 * c);
}

// ---------------------------------------------------------------------------
// Causal flash attention. Grid (S/64, H, B), 128 threads = 4 waves.
// Wave w owns q rows [qb + 16w, qb + 16w + 16). KV block = 32 rows.
// ---------------------------------------------------------------------------
__global__ __launch_bounds__(128) void flash_attn(const _Float16* __restrict__ Q,
                                                  const _Float16* __restrict__ K,
                                                  const _Float16* __restrict__ V,
                                                  _Float16* __restrict__ O) {
  __shared__ _Float16 sVt[128 * 40];   // V^T tile: [d=128][kv=32] padded
  __shared__ _Float16 sP[4][16 * 40];  // per-wave P staging (16x32 padded)
  const int b = blockIdx.z, h = blockIdx.y, qb = blockIdx.x * 64;
  const int tid = threadIdx.x, wid = tid >> 5, lane = tid & 31;
  const int ln = lane & 15, hi = lane >> 4;
  const float scale = 0.08838834764831845f;  // 1/sqrt(128)
  const long S = 2048, D = 2048;
  const long bh = (long)b * S * D + (long)h * 128;

  // Q A-fragments for this wave's 16 rows (4 chunks of K=32 over head_dim=128)
  v16h qa[4];
  {
    const _Float16* qrow = Q + bh + (long)(qb + wid * 16 + ln) * D;
#pragma unroll
    for (int c = 0; c < 4; ++c) {
      const _Float16* p = qrow + c * 32;
      qa[c] = cat8(*(const v8h*)(p + hi * 8), *(const v8h*)(p + 16 + hi * 8));
    }
  }

  const v8f vzero = {0.f, 0.f, 0.f, 0.f, 0.f, 0.f, 0.f, 0.f};
  v8f o[8];
  float mrun[8], lrun[8];
#pragma unroll
  for (int t = 0; t < 8; ++t) { o[t] = vzero; mrun[t] = -__builtin_inff(); lrun[t] = 0.f; }

  const int nblk = (qb + 64) >> 5;  // uniform across the block -> safe barriers
  for (int j = 0; j < nblk; ++j) {
    const int jb = j * 32;
    __syncthreads();
    {  // stage V transposed: sVt[d][kv]
      const int r = tid >> 2, cc = (tid & 3) * 32;
      const _Float16* vp = V + bh + (long)(jb + r) * D + cc;
#pragma unroll
      for (int i = 0; i < 32; ++i) sVt[(cc + i) * 40 + r] = vp[i];
    }
    __syncthreads();

    // S = Q K^T : two 16x16 N-tiles, inner dim 128 = 4 WMMA chains
    v8f sc[2];
#pragma unroll
    for (int t = 0; t < 2; ++t) {
      v8f a = vzero;
      const _Float16* krow = K + bh + (long)(jb + t * 16 + ln) * D;
#pragma unroll
      for (int c = 0; c < 4; ++c) {
        const _Float16* p = krow + c * 32;
        v16h kb = cat8(*(const v8h*)(p + hi * 8), *(const v8h*)(p + 16 + hi * 8));
        a = WMMA_F16(qa[c], kb, a);
      }
      sc[t] = a;
    }

    // causal mask + scale + online softmax (row stats live per vgpr slot)
    float rm[8];
#pragma unroll
    for (int v = 0; v < 8; ++v) {
      const int q = qb + wid * 16 + hi * 8 + v;
      float s0 = (jb + ln <= q) ? sc[0][v] * scale : -__builtin_inff();
      float s1 = (jb + 16 + ln <= q) ? sc[1][v] * scale : -__builtin_inff();
      sc[0][v] = s0; sc[1][v] = s1;
      rm[v] = fmaxf(s0, s1);
    }
    float alpha[8];
#pragma unroll
    for (int v = 0; v < 8; ++v) {
#pragma unroll
      for (int x = 1; x <= 8; x <<= 1) rm[v] = fmaxf(rm[v], __shfl_xor(rm[v], x, 32));
      const float mn = fmaxf(mrun[v], rm[v]);
      alpha[v] = __expf(mrun[v] - mn);
      mrun[v] = mn;
      const float p0 = __expf(sc[0][v] - mn);
      const float p1 = __expf(sc[1][v] - mn);
      sc[0][v] = p0; sc[1][v] = p1;
      float r = p0 + p1;
#pragma unroll
      for (int x = 1; x <= 8; x <<= 1) r += __shfl_xor(r, x, 32);
      lrun[v] = lrun[v] * alpha[v] + r;
    }
#pragma unroll
    for (int dt = 0; dt < 8; ++dt)
#pragma unroll
      for (int v = 0; v < 8; ++v) o[dt][v] *= alpha[v];

    // stage P (C layout -> row-major LDS) for A-fragment reuse; wave-local fence
    _Float16* pb = &sP[wid][0];
#pragma unroll
    for (int v = 0; v < 8; ++v) {
      const int m = hi * 8 + v;
      pb[m * 40 + ln]      = (_Float16)sc[0][v];
      pb[m * 40 + 16 + ln] = (_Float16)sc[1][v];
    }
    asm volatile("s_wait_dscnt 0" ::: "memory");
    __builtin_amdgcn_wave_barrier();

    // O += P * V  (inner dim = 32 kv rows; V^T rows are K-contiguous)
    v16h pa = fragA(pb, 40);
#pragma unroll
    for (int dt = 0; dt < 8; ++dt) {
      v16h vb = fragB(&sVt[dt * 16 * 40], 40);
      o[dt] = WMMA_F16(pa, vb, o[dt]);
    }
  }

  // normalize + write f16 output (layout [B,S,H*128])
#pragma unroll
  for (int v = 0; v < 8; ++v) {
    const float inv = 1.0f / lrun[v];
    const long row = bh + (long)(qb + wid * 16 + hi * 8 + v) * D;
#pragma unroll
    for (int dt = 0; dt < 8; ++dt)
      O[row + dt * 16 + ln] = (_Float16)(o[dt][v] * inv);
  }
}

// ---------------------------------------------------------------------------
// Host side
// ---------------------------------------------------------------------------
extern "C" void kernel_launch(void* const* d_in, const int* in_sizes, int n_in,
                              void* d_out, int out_size, void* d_ws, size_t ws_size,
                              hipStream_t stream) {
  (void)in_sizes; (void)n_in; (void)out_size; (void)ws_size;
  const float* x    = (const float*)d_in[0];
  const float* wq   = (const float*)d_in[1];
  const float* wk   = (const float*)d_in[2];
  const float* wv   = (const float*)d_in[3];
  const float* wo   = (const float*)d_in[4];
  const float* cosc = (const float*)d_in[5];
  const float* sinc = (const float*)d_in[6];
  const int*   offp = (const int*)d_in[8];   // d_in[7] = mask (causal, computed in-kernel)
  float* out = (float*)d_out;

  const int NX = 8388608;  // B*S*D
  const int NW = 4194304;  // D*D
  char* ws = (char*)d_ws;
  _Float16* xh  = (_Float16*)(ws + 0);
  _Float16* wqh = (_Float16*)(ws + 16777216);
  _Float16* wkh = (_Float16*)(ws + 25165824);
  _Float16* wvh = (_Float16*)(ws + 33554432);
  _Float16* woh = (_Float16*)(ws + 41943040);
  _Float16* Qh  = (_Float16*)(ws + 50331648);
  _Float16* Kh  = (_Float16*)(ws + 67108864);
  _Float16* Vh  = (_Float16*)(ws + 83886080);
  _Float16* Oh  = (_Float16*)(ws + 100663296);

  cvt_f32_f16<<<NX / 1024, 256, 0, stream>>>(x, xh, NX);
  cvt_f32_f16<<<NW / 1024, 256, 0, stream>>>(wq, wqh, NW);
  cvt_f32_f16<<<NW / 1024, 256, 0, stream>>>(wk, wkh, NW);
  cvt_f32_f16<<<NW / 1024, 256, 0, stream>>>(wv, wvh, NW);
  cvt_f32_f16<<<NW / 1024, 256, 0, stream>>>(wo, woh, NW);

  dim3 gg(16, 32);  // N/128, M/128  (M = B*S = 4096, N = D = 2048)
  gemm_nt<_Float16><<<gg, 256, 0, stream>>>(xh, wqh, Qh, 4096, 2048, 2048);
  gemm_nt<_Float16><<<gg, 256, 0, stream>>>(xh, wkh, Kh, 4096, 2048, 2048);
  gemm_nt<_Float16><<<gg, 256, 0, stream>>>(xh, wvh, Vh, 4096, 2048, 2048);

  rope_qk<<<4194304 / 256, 256, 0, stream>>>(Qh, Kh, cosc, sinc, offp);

  dim3 ga(32, 16, 2);  // S/64, H, B
  flash_attn<<<ga, 128, 0, stream>>>(Qh, Kh, Vh, Oh);

  gemm_nt<float><<<gg, 256, 0, stream>>>(Oh, woh, out, 4096, 2048, 2048);
}
